// SS2Dv0_12713103196343
// MI455X (gfx1250) — compile-verified, matrix operand order checked by reading, not run on hardware
//
#include <hip/hip_runtime.h>
#include <hip/hip_bf16.h>
#include <math.h>

#define BB   2
#define LL   4096
#define DMM  64
#define DII  128
#define KK   4
#define NSTN 16
#define RR   4
#define CC   36   // R + 2*NST

typedef __attribute__((ext_vector_type(16))) __bf16 bf16x16;
typedef __attribute__((ext_vector_type(8)))  float  floatx8;

// K-index of element i of a 16-bit A/B WMMA fragment (16x32), per ISA 7.12.2
__device__ __forceinline__ int kofs(int lane, int i) {
  int half = (lane >> 4) & 1;
  int p = i >> 1;
  return ((p >> 2) << 4) + (half << 3) + ((p & 3) << 1) + (i & 1);
}

// Load one 16x32-bf16 fragment row/col slice from row-major f32 memory.
// Elements 0..7  -> k = half*8 + 0..7      (contiguous)
// Elements 8..15 -> k = 16 + half*8 + 0..7 (contiguous)
__device__ __forceinline__ bf16x16 ld_frag32(const float* __restrict__ p, int half) {
  const float4* q0 = (const float4*)(p + (half << 3));
  const float4* q1 = (const float4*)(p + 16 + (half << 3));
  float4 a = q0[0], b = q0[1], c = q1[0], d = q1[1];
  bf16x16 r;
  r[0]  = (__bf16)a.x; r[1]  = (__bf16)a.y; r[2]  = (__bf16)a.z; r[3]  = (__bf16)a.w;
  r[4]  = (__bf16)b.x; r[5]  = (__bf16)b.y; r[6]  = (__bf16)b.z; r[7]  = (__bf16)b.w;
  r[8]  = (__bf16)c.x; r[9]  = (__bf16)c.y; r[10] = (__bf16)c.z; r[11] = (__bf16)c.w;
  r[12] = (__bf16)d.x; r[13] = (__bf16)d.y; r[14] = (__bf16)d.z; r[15] = (__bf16)d.w;
  return r;
}

// silu via single v_rcp_f32 (avoid IEEE div expansion)
__device__ __forceinline__ float siluf(float x) {
  return x * __builtin_amdgcn_rcpf(1.f + __expf(-x));
}
// softplus via v_exp_f32 + v_log_f32 (avoid ocml log1p call on the scan critical path)
__device__ __forceinline__ float softplusf(float x) {
  return (x > 20.f) ? x : __logf(1.f + __expf(x));
}

// scan position l of direction k -> spatial index (row-major hw); same map for gather/scatter
__device__ __forceinline__ int mapidx(int k, int l) {
  switch (k) {
    case 0:  return l;
    case 1:  return ((l & 63) << 6) + (l >> 6);
    case 2:  return (LL - 1) - l;
    default: { int l2 = (LL - 1) - l; return ((l2 & 63) << 6) + (l2 >> 6); }
  }
}

// ---------------- in_proj: (B*L,64) x (64,256)^T -> xi (..,128), silu(z) (..,128)
__global__ void k_inproj(const float* __restrict__ x, const float* __restrict__ w,
                         float* __restrict__ xi, float* __restrict__ zb) {
  int lane = threadIdx.x & 31;
  int wave = threadIdx.x >> 5;
  int half = lane >> 4;
  int tile = blockIdx.x * 8 + wave;       // 8192 tiles: tm 0..511, tn 0..15
  int tm = tile >> 4;
  int tn = tile & 15;
  int row  = tm * 16 + (lane & 15);
  int ncol = tn * 16 + (lane & 15);
  bf16x16 a0 = ld_frag32(x + (size_t)row * 64, half);
  bf16x16 a1 = ld_frag32(x + (size_t)row * 64 + 32, half);
  bf16x16 b0 = ld_frag32(w + (size_t)ncol * 64, half);
  bf16x16 b1 = ld_frag32(w + (size_t)ncol * 64 + 32, half);
  floatx8 c = {};
  c = __builtin_amdgcn_wmma_f32_16x16x32_bf16(false, a0, false, b0, (short)0, c, false, false);
  c = __builtin_amdgcn_wmma_f32_16x16x32_bf16(false, a1, false, b1, (short)0, c, false, false);
  int mb = tm * 16 + (half << 3);
  #pragma unroll
  for (int v = 0; v < 8; ++v) {
    int m = mb + v;
    if (ncol < DII) xi[(size_t)m * DII + ncol] = c[v];
    else            zb[(size_t)m * DII + (ncol - DII)] = siluf(c[v]);
  }
}

// ---------------- depthwise 3x3 conv + bias + silu; xi (b,l,di) -> xc (b,di,l)
__global__ void k_conv(const float* __restrict__ xi, const float* __restrict__ cw,
                       const float* __restrict__ cb, float* __restrict__ xc) {
  int idx = blockIdx.x * blockDim.x + threadIdx.x;   // ((b*128+d)*4096)+h*64+w
  if (idx >= BB * DII * LL) return;
  int w = idx & 63, h = (idx >> 6) & 63, d = (idx >> 12) & 127, b = idx >> 19;
  float acc = cb[d];
  #pragma unroll
  for (int dy = 0; dy < 3; ++dy) {
    int hh = h + dy - 1; if (hh < 0 || hh >= 64) continue;
    #pragma unroll
    for (int dx = 0; dx < 3; ++dx) {
      int ww = w + dx - 1; if (ww < 0 || ww >= 64) continue;
      acc += xi[(size_t)((b * LL) + hh * 64 + ww) * DII + d] * cw[d * 9 + dy * 3 + dx];
    }
  }
  xc[(size_t)(b * DII + d) * LL + h * 64 + w] = siluf(acc);
}

// ---------------- x_dbl: per (b,k): (36x128 padded to 48) x (128 x L) via WMMA
__global__ void k_xdbl(const float* __restrict__ xc, const float* __restrict__ wp,
                       float* __restrict__ xdbl) {
  int lane = threadIdx.x & 31;
  int wave = threadIdx.x >> 5;
  int half = lane >> 4;
  int tile = blockIdx.x * 8 + wave;       // 6144 tiles = 8 bk * 3 mtiles * 256 ltiles
  int bk = tile / (3 * 256);
  int rest = tile % (3 * 256);
  int tmi = rest >> 8;
  int tl = rest & 255;
  int b = bk >> 2, k = bk & 3;
  int crow = tmi * 16 + (lane & 15);
  int l = tl * 16 + (lane & 15);
  int src = mapidx(k, l);
  floatx8 c = {};
  #pragma unroll
  for (int ks = 0; ks < 4; ++ks) {
    bf16x16 a, bb;
    if (crow < CC) {
      a = ld_frag32(wp + (size_t)(k * CC + crow) * DII + ks * 32, half);
    } else {
      #pragma unroll
      for (int i = 0; i < 16; ++i) a[i] = (__bf16)0.f;
    }
    #pragma unroll
    for (int i = 0; i < 16; ++i) {
      int kd = ks * 32 + kofs(lane, i);
      bb[i] = (__bf16)xc[(size_t)(b * DII + kd) * LL + src];
    }
    c = __builtin_amdgcn_wmma_f32_16x16x32_bf16(false, a, false, bb, (short)0, c, false, false);
  }
  int cb0 = tmi * 16 + (half << 3);
  #pragma unroll
  for (int v = 0; v < 8; ++v) {
    int cr = cb0 + v;
    if (cr < CC) xdbl[(size_t)(bk * CC + cr) * LL + l] = c[v];
  }
}

// ---------------- selective scan: one block per (b,k); lane = channel d; 16 states in VGPRs
__global__ void k_scan(const float* __restrict__ xc, const float* __restrict__ xdbl,
                       const float* __restrict__ dtw, const float* __restrict__ dtb,
                       const float* __restrict__ alogs, const float* __restrict__ ds,
                       float* __restrict__ ysum) {
  int bk = blockIdx.x, b = bk >> 2, k = bk & 3;
  int d = threadIdx.x;               // 0..127
  float A[NSTN];
  #pragma unroll
  for (int n = 0; n < NSTN; ++n) A[n] = -__expf(alogs[(size_t)(k * DII + d) * NSTN + n]);
  float Dd   = ds[k * DII + d];
  float bias = dtb[k * DII + d];
  float w0 = dtw[(size_t)(k * DII + d) * RR + 0];
  float w1 = dtw[(size_t)(k * DII + d) * RR + 1];
  float w2 = dtw[(size_t)(k * DII + d) * RR + 2];
  float w3 = dtw[(size_t)(k * DII + d) * RR + 3];
  float st[NSTN];
  #pragma unroll
  for (int n = 0; n < NSTN; ++n) st[n] = 0.f;

  __shared__ float sc[2][CC];
  const float* xd  = xdbl + (size_t)bk * CC * LL;
  const float* xcb = xc + (size_t)b * DII * LL + (size_t)d * LL;

  for (int l = 0; l < LL; ++l) {
    int buf = l & 1;
    if (d < CC) {
      sc[buf][d] = xd[(size_t)d * LL + l];
      if (l + 16 < LL) __builtin_prefetch(&xd[(size_t)d * LL + l + 16], 0, 0);
    }
    __syncthreads();
    int s = mapidx(k, l);
    float u = xcb[s];
    float delta = softplusf(bias + sc[buf][0] * w0 + sc[buf][1] * w1 +
                                   sc[buf][2] * w2 + sc[buf][3] * w3);
    float du = delta * u;
    float y = 0.f;
    #pragma unroll
    for (int n = 0; n < NSTN; ++n) {
      float dA = __expf(delta * A[n]);
      st[n] = st[n] * dA + du * sc[buf][RR + n];
      y += st[n] * sc[buf][RR + NSTN + n];
    }
    y += u * Dd;
    atomicAdd(&ysum[(size_t)(b * LL + s) * DII + d], y);
  }
}

// ---------------- LayerNorm(DI) * silu(z), in place on ysum
__global__ void k_ln(float* __restrict__ y, const float* __restrict__ zb,
                     const float* __restrict__ g, const float* __restrict__ be) {
  int r = blockIdx.x, t = threadIdx.x;
  float v = y[(size_t)r * DII + t];
  __shared__ float s1[DII], s2[DII];
  s1[t] = v; s2[t] = v * v;
  __syncthreads();
  for (int s = 64; s > 0; s >>= 1) {
    if (t < s) { s1[t] += s1[t + s]; s2[t] += s2[t + s]; }
    __syncthreads();
  }
  float mean = s1[0] * (1.f / DII);
  float var  = s2[0] * (1.f / DII) - mean * mean;
  float yh = (v - mean) * rsqrtf(var + 1e-5f) * g[t] + be[t];
  y[(size_t)r * DII + t] = yh * zb[(size_t)r * DII + t];
}

// ---------------- out_proj: (B*L,128) x (128,64)^T via WMMA
__global__ void k_outproj(const float* __restrict__ y, const float* __restrict__ w,
                          float* __restrict__ out) {
  int lane = threadIdx.x & 31;
  int wave = threadIdx.x >> 5;
  int half = lane >> 4;
  int tile = blockIdx.x * 8 + wave;    // 2048 tiles: tm 0..511, tn 0..3
  int tm = tile >> 2;
  int tn = tile & 3;
  int row = tm * 16 + (lane & 15);
  int n = tn * 16 + (lane & 15);
  floatx8 c = {};
  #pragma unroll
  for (int ks = 0; ks < 4; ++ks) {
    bf16x16 a  = ld_frag32(y + (size_t)row * DII + ks * 32, half);
    bf16x16 bb = ld_frag32(w + (size_t)n * DII + ks * 32, half);
    c = __builtin_amdgcn_wmma_f32_16x16x32_bf16(false, a, false, bb, (short)0, c, false, false);
  }
  int mb = tm * 16 + (half << 3);
  #pragma unroll
  for (int v = 0; v < 8; ++v)
    out[(size_t)(mb + v) * DMM + (tn * 16 + (lane & 15))] = c[v];
}

extern "C" void kernel_launch(void* const* d_in, const int* in_sizes, int n_in,
                              void* d_out, int out_size, void* d_ws, size_t ws_size,
                              hipStream_t stream) {
  (void)in_sizes; (void)n_in; (void)out_size; (void)ws_size;
  const float* x     = (const float*)d_in[0];
  const float* ipw   = (const float*)d_in[1];
  const float* cw    = (const float*)d_in[2];
  const float* cb    = (const float*)d_in[3];
  const float* xpw   = (const float*)d_in[4];
  const float* dtw   = (const float*)d_in[5];
  const float* dtb   = (const float*)d_in[6];
  const float* alogs = (const float*)d_in[7];
  const float* ds    = (const float*)d_in[8];
  const float* g     = (const float*)d_in[9];
  const float* be    = (const float*)d_in[10];
  const float* opw   = (const float*)d_in[11];
  float* out = (float*)d_out;

  float* ws   = (float*)d_ws;
  float* xi   = ws;                 // (b,l,128)   1048576
  float* zb   = xi + 1048576;       // (b,l,128)   1048576
  float* xc   = zb + 1048576;       // (b,128,l)   1048576
  float* xdbl = xc + 1048576;       // (b,k,36,l)  1179648
  float* ysum = xdbl + 1179648;     // (b,l,128)   1048576

  hipMemsetAsync(ysum, 0, 1048576 * sizeof(float), stream);
  k_inproj <<<1024, 256, 0, stream>>>(x, ipw, xi, zb);
  k_conv   <<<4096, 256, 0, stream>>>(xi, cw, cb, xc);
  k_xdbl   <<<768,  256, 0, stream>>>(xc, xpw, xdbl);
  k_scan   <<<8,    128, 0, stream>>>(xc, xdbl, dtw, dtb, alogs, ds, ysum);
  k_ln     <<<8192, 128, 0, stream>>>(ysum, zb, g, be);
  k_outproj<<<256,  256, 0, stream>>>(ysum, opw, out);
}
